// MultiHeadAttention_13520557048510
// MI455X (gfx1250) — matrix operand order repaired and checked by reading, not compiled, not run on hardware
//
#include <hip/hip_runtime.h>
#include <hip/hip_bf16.h>

#define EMBED    1024
#define HEADS    16
#define HEAD_DIM 64
#define BATCH    4
#define SEQ      2048
#define MROWS    (BATCH * SEQ)

typedef __bf16 bf16_t;
typedef __attribute__((ext_vector_type(16))) __bf16 v16bf;
typedef __attribute__((ext_vector_type(8)))  __bf16 v8bf;
typedef __attribute__((ext_vector_type(8)))  float  v8f;

// -------------------------------------------------------------------------
// A-fragment for V_WMMA_F32_16X16X32_BF16 (16x32, 16-bit A layout, wave32):
//   lane L holds row m = L&15; element j holds K = 16*(L>>4) + j (this code's contiguous k order, B matches)
// `base` must point at  rowPtr + k + 16*(lane>>4)  (two adjacent 16B loads).
// -------------------------------------------------------------------------
__device__ __forceinline__ v16bf make_a_frag(const bf16_t* base) {
  v8bf lo = *(const v8bf*)(base);
  v8bf hi = *(const v8bf*)(base + 8);
  v16bf a;
#pragma unroll
  for (int j = 0; j < 8; ++j) { a[j] = lo[j]; a[j + 8] = hi[j]; }
  return a;
}

// -------------------------------------------------------------------------
// Elementwise f32 -> bf16 pack
// -------------------------------------------------------------------------
__global__ void pack_x_bf16(const float* __restrict__ X, bf16_t* __restrict__ Xb, int n) {
  int i = blockIdx.x * blockDim.x + threadIdx.x;
  if (i < n) Xb[i] = (bf16_t)X[i];
}

// -------------------------------------------------------------------------
// Transpose-pack weights: W f32 [K=E][N=E] row-major -> Wt bf16 [N][K]
// -------------------------------------------------------------------------
__global__ void pack_wT_bf16(const float* __restrict__ W, bf16_t* __restrict__ Wt) {
  __shared__ float tile[16][17];
  int n = blockIdx.x * 16 + threadIdx.x;
  int k = blockIdx.y * 16 + threadIdx.y;
  tile[threadIdx.y][threadIdx.x] = W[(size_t)k * EMBED + n];
  __syncthreads();
  int n2 = blockIdx.x * 16 + threadIdx.y;
  int k2 = blockIdx.y * 16 + threadIdx.x;
  Wt[(size_t)n2 * EMBED + k2] = (bf16_t)tile[threadIdx.x][threadIdx.y];
}

// -------------------------------------------------------------------------
// bf16 WMMA GEMM, register-blocked: each wave computes a 64x32 C macro-tile
// (4 M-subtiles x 2 N-subtiles) -> 8 WMMAs per 12 b128 loads per K-step.
// Block = 4 waves arranged 2x2 -> 128x64 C per block.
// mode 0: Q  -> bf16 [B,H,S,D], scaled by alpha = 1/sqrt(D)
// mode 1: K  -> bf16 [B,H,S,D]
// mode 2: V  -> bf16 [B,H,D,S]   (transposed for contiguous PV B-fragments)
// mode 3: f32 row-major [M,N]    (final output projection)
// -------------------------------------------------------------------------
__global__ void gemm_bf16_wmma(const bf16_t* __restrict__ A,
                               const bf16_t* __restrict__ Bt,
                               const float*  __restrict__ bias,
                               void* __restrict__ Cout,
                               int M, int N, int K,
                               int mode, float alpha) {
  const int wave = threadIdx.x >> 5;
  const int lane = threadIdx.x & 31;
  const int hi   = lane >> 4;
  const int ln   = lane & 15;
  const int m0   = blockIdx.x * 128 + (wave >> 1) * 64;  // rows m0..m0+63
  const int n0   = blockIdx.y * 64  + (wave & 1) * 32;   // cols n0..n0+31

  const bf16_t* arow[4];
  const bf16_t* brow[2];
#pragma unroll
  for (int i = 0; i < 4; ++i) arow[i] = A + (size_t)(m0 + i * 16 + ln) * K;
#pragma unroll
  for (int j = 0; j < 2; ++j) brow[j] = Bt + (size_t)(n0 + j * 16 + ln) * K;

  v8f c[4][2];
  v8f zero = {};
#pragma unroll
  for (int i = 0; i < 4; ++i)
#pragma unroll
    for (int j = 0; j < 2; ++j) c[i][j] = zero;

  for (int k = 0; k < K; k += 32) {
    v16bf a[4], b[2];
#pragma unroll
    for (int i = 0; i < 4; ++i) {
      __builtin_prefetch(arow[i] + k + 256);             // global_prefetch_b8
      a[i] = make_a_frag(arow[i] + k + hi * 16);
    }
#pragma unroll
    for (int j = 0; j < 2; ++j) {
      __builtin_prefetch(brow[j] + k + 256);
      b[j] = *(const v16bf*)(brow[j] + k + hi * 16);     // K = hi*16 + jj, contiguous
    }
#pragma unroll
    for (int i = 0; i < 4; ++i)
#pragma unroll
      for (int j = 0; j < 2; ++j)
        c[i][j] = __builtin_amdgcn_wmma_f32_16x16x32_bf16(false, a[i], false, b[j],
                                                          (short)0, c[i][j], false, false);
  }

#pragma unroll
  for (int j = 0; j < 2; ++j) {
    const int gc   = n0 + j * 16 + ln;
    const float bv = bias[gc];
    if (mode == 3) {
      float* O = (float*)Cout;
#pragma unroll
      for (int i = 0; i < 4; ++i)
#pragma unroll
        for (int r = 0; r < 8; ++r) {
          int gm = m0 + i * 16 + r + hi * 8;             // C layout: row = r + 8*hi
          O[(size_t)gm * N + gc] = c[i][j][r] + bv;
        }
    } else {
      bf16_t* O  = (bf16_t*)Cout;
      const int h = gc >> 6;        // / HEAD_DIM
      const int d = gc & 63;
#pragma unroll
      for (int i = 0; i < 4; ++i)
#pragma unroll
        for (int r = 0; r < 8; ++r) {
          int gm = m0 + i * 16 + r + hi * 8;
          int bb = gm >> 11;        // / SEQ
          int ss = gm & (SEQ - 1);
          float v = (c[i][j][r] + bv) * alpha;
          if (mode == 2)
            O[(((size_t)bb * HEADS + h) * HEAD_DIM + d) * SEQ + ss] = (bf16_t)v;
          else
            O[(((size_t)bb * HEADS + h) * SEQ + ss) * HEAD_DIM + d] = (bf16_t)v;
        }
    }
  }
}

// -------------------------------------------------------------------------
// Flash attention: 32 queries per wave (two 16-row tiles sharing all K/V
// B-fragments), streaming 32-key tiles. Q,K bf16 [B,H,S,D] (Q pre-scaled),
// Vt bf16 [B,H,D,S]. f32 online softmax; P goes through LDS (C->A layout).
// Output O: bf16 [B*S, E] row-major (already (b, s, h*D+d) order).
// -------------------------------------------------------------------------
__global__ void attn_fwd_wmma(const bf16_t* __restrict__ Q,
                              const bf16_t* __restrict__ Kt,
                              const bf16_t* __restrict__ Vt,
                              bf16_t* __restrict__ O) {
  __shared__ __align__(16) bf16_t pshare[4 * 2 * 16 * 32];   // 8 KB: 2 P-tiles/wave
  const int wave = threadIdx.x >> 5;
  const int lane = threadIdx.x & 31;
  const int hi   = lane >> 4;
  const int ln   = lane & 15;
  const int bh   = blockIdx.x;                 // b*HEADS + h
  const int b    = bh >> 4;
  const int h    = bh & 15;
  const int q0   = (blockIdx.y * 4 + wave) * 32;
  bf16_t* pb     = pshare + wave * (2 * 16 * 32);

  // Q A-fragments: 2 query subtiles x 2 d-halves
  v16bf aq[2][2];
#pragma unroll
  for (int qt = 0; qt < 2; ++qt) {
    const bf16_t* Qbase = Q + ((size_t)bh * SEQ + q0 + qt * 16 + ln) * HEAD_DIM;
    aq[qt][0] = make_a_frag(Qbase + hi * 16);
    aq[qt][1] = make_a_frag(Qbase + 32 + hi * 16);
  }

  float mrow[2][8], lrow[2][8];
  v8f acc[2][4];
  v8f zero = {};
#pragma unroll
  for (int qt = 0; qt < 2; ++qt) {
#pragma unroll
    for (int r = 0; r < 8; ++r) { mrow[qt][r] = -3.0e38f; lrow[qt][r] = 0.0f; }
#pragma unroll
    for (int t = 0; t < 4; ++t) acc[qt][t] = zero;
  }

  const bf16_t* Kbh = Kt + (size_t)bh * SEQ * HEAD_DIM;
  const bf16_t* Vbh = Vt + (size_t)bh * HEAD_DIM * SEQ;

  for (int kt = 0; kt < SEQ; kt += 32) {
    const bf16_t* kp0 = Kbh + (size_t)(kt + ln) * HEAD_DIM;       // keys kt+0..15
    const bf16_t* kp1 = Kbh + (size_t)(kt + 16 + ln) * HEAD_DIM;  // keys kt+16..31
    __builtin_prefetch(kp0 + 32 * HEAD_DIM);
    __builtin_prefetch(kp1 + 32 * HEAD_DIM);

    // --- scores: keep only 2 K-fragments live at a time ---
    v8f c0[2], c1[2];
    {
      v16bf bk00 = *(const v16bf*)(kp0 + hi * 16);                // keys lo, d 0..31
      v16bf bk10 = *(const v16bf*)(kp1 + hi * 16);                // keys hi, d 0..31
#pragma unroll
      for (int qt = 0; qt < 2; ++qt) {
        c0[qt] = __builtin_amdgcn_wmma_f32_16x16x32_bf16(false, aq[qt][0], false, bk00,
                                                         (short)0, zero, false, false);
        c1[qt] = __builtin_amdgcn_wmma_f32_16x16x32_bf16(false, aq[qt][0], false, bk10,
                                                         (short)0, zero, false, false);
      }
    }
    {
      v16bf bk01 = *(const v16bf*)(kp0 + 32 + hi * 16);           // keys lo, d 32..63
      v16bf bk11 = *(const v16bf*)(kp1 + 32 + hi * 16);           // keys hi, d 32..63
#pragma unroll
      for (int qt = 0; qt < 2; ++qt) {
        c0[qt] = __builtin_amdgcn_wmma_f32_16x16x32_bf16(false, aq[qt][1], false, bk01,
                                                         (short)0, c0[qt], false, false);
        c1[qt] = __builtin_amdgcn_wmma_f32_16x16x32_bf16(false, aq[qt][1], false, bk11,
                                                         (short)0, c1[qt], false, false);
      }
    }

    // --- online softmax per query subtile (f32).  xor masks 1/2/4/8 reduce
    //     over the 16-lane column group without crossing wave32 halves. ---
#pragma unroll
    for (int qt = 0; qt < 2; ++qt) {
      bf16_t* pq = pb + qt * (16 * 32);
#pragma unroll
      for (int r = 0; r < 8; ++r) {
        float s0 = c0[qt][r], s1 = c1[qt][r];
        float mx = fmaxf(s0, s1);
#pragma unroll
        for (int off = 1; off < 16; off <<= 1)
          mx = fmaxf(mx, __shfl_xor(mx, off));
        float mnew  = fmaxf(mrow[qt][r], mx);
        float scale = __expf(mrow[qt][r] - mnew);
        float p0 = __expf(s0 - mnew);
        float p1 = __expf(s1 - mnew);
        float ps = p0 + p1;
#pragma unroll
        for (int off = 1; off < 16; off <<= 1)
          ps += __shfl_xor(ps, off);
        lrow[qt][r] = lrow[qt][r] * scale + ps;
        mrow[qt][r] = mnew;
#pragma unroll
        for (int t = 0; t < 4; ++t) acc[qt][t][r] = acc[qt][t][r] * scale;
        pq[(r + hi * 8) * 32 + ln]      = (bf16_t)p0;   // C-layout coords
        pq[(r + hi * 8) * 32 + 16 + ln] = (bf16_t)p1;
      }
    }

    // --- reload P tiles as 16x32 A-fragments (same wave: LDS in-order) ---
    v16bf aP[2];
#pragma unroll
    for (int qt = 0; qt < 2; ++qt) {
      const bf16_t* pq = pb + qt * (16 * 32);
      v8bf plo = *(const v8bf*)(pq + ln * 32 + hi * 16);
      v8bf phi = *(const v8bf*)(pq + ln * 32 + 8 + hi * 16);
#pragma unroll
      for (int j = 0; j < 8; ++j) { aP[qt][j] = plo[j]; aP[qt][j + 8] = phi[j]; }
    }

    // --- O += P * V : each V B-fragment feeds both query subtiles ---
#pragma unroll
    for (int t = 0; t < 4; ++t) {
      const bf16_t* vp = Vbh + (size_t)(t * 16 + ln) * SEQ + kt + hi * 16;
      v16bf bv = *(const v16bf*)vp;
#pragma unroll
      for (int qt = 0; qt < 2; ++qt)
        acc[qt][t] = __builtin_amdgcn_wmma_f32_16x16x32_bf16(false, aP[qt], false, bv,
                                                             (short)0, acc[qt][t], false, false);
    }
  }

  // epilogue: normalize by l and store [b, s, h*D + d]
#pragma unroll
  for (int qt = 0; qt < 2; ++qt)
#pragma unroll
    for (int t = 0; t < 4; ++t)
#pragma unroll
      for (int r = 0; r < 8; ++r) {
        float v  = acc[qt][t][r] / lrow[qt][r];
        int  ss  = q0 + qt * 16 + r + hi * 8;
        int  col = h * HEAD_DIM + t * 16 + ln;
        O[((size_t)b * SEQ + ss) * EMBED + col] = (bf16_t)v;
      }
}

// -------------------------------------------------------------------------
extern "C" void kernel_launch(void* const* d_in, const int* in_sizes, int n_in,
                              void* d_out, int out_size, void* d_ws, size_t ws_size,
                              hipStream_t stream) {
  const float* x  = (const float*)d_in[0];
  const float* Wq = (const float*)d_in[1];
  const float* bq = (const float*)d_in[2];
  const float* Wk = (const float*)d_in[3];
  const float* bk = (const float*)d_in[4];
  const float* Wv = (const float*)d_in[5];
  const float* bv = (const float*)d_in[6];
  const float* Wo = (const float*)d_in[7];
  const float* bo = (const float*)d_in[8];
  float* out = (float*)d_out;

  char* ws = (char*)d_ws;
  const size_t MB = 1024ull * 1024ull;
  bf16_t* xb  = (bf16_t*)(ws + 0);        // 16 MB  x in bf16
  bf16_t* Wqt = (bf16_t*)(ws + 16 * MB);  //  2 MB  each, transposed bf16
  bf16_t* Wkt = (bf16_t*)(ws + 18 * MB);
  bf16_t* Wvt = (bf16_t*)(ws + 20 * MB);
  bf16_t* Wot = (bf16_t*)(ws + 22 * MB);
  bf16_t* Qh  = (bf16_t*)(ws + 24 * MB);  // 16 MB  [B,H,S,D] (pre-scaled)
  bf16_t* Kh  = (bf16_t*)(ws + 40 * MB);  // 16 MB  [B,H,S,D]
  bf16_t* Vth = (bf16_t*)(ws + 56 * MB);  // 16 MB  [B,H,D,S]
  bf16_t* AO  = (bf16_t*)(ws + 72 * MB);  // 16 MB  attention output [B*S,E]

  // 1) pack inputs to bf16 (weights transposed)
  pack_x_bf16<<<(MROWS * EMBED) / 256, 256, 0, stream>>>(x, xb, MROWS * EMBED);
  dim3 tgrid(EMBED / 16, EMBED / 16), tblk(16, 16);
  pack_wT_bf16<<<tgrid, tblk, 0, stream>>>(Wq, Wqt);
  pack_wT_bf16<<<tgrid, tblk, 0, stream>>>(Wk, Wkt);
  pack_wT_bf16<<<tgrid, tblk, 0, stream>>>(Wv, Wvt);
  pack_wT_bf16<<<tgrid, tblk, 0, stream>>>(Wo, Wot);

  // 2) QKV projections (Q scaled by 1/sqrt(D); V stored transposed)
  dim3 ggrid(MROWS / 128, EMBED / 64), gblk(128);
  gemm_bf16_wmma<<<ggrid, gblk, 0, stream>>>(xb, Wqt, bq, Qh,  MROWS, EMBED, EMBED, 0, 0.125f);
  gemm_bf16_wmma<<<ggrid, gblk, 0, stream>>>(xb, Wkt, bk, Kh,  MROWS, EMBED, EMBED, 1, 1.0f);
  gemm_bf16_wmma<<<ggrid, gblk, 0, stream>>>(xb, Wvt, bv, Vth, MROWS, EMBED, EMBED, 2, 1.0f);

  // 3) flash attention (32 queries/wave, 128/block)
  dim3 agrid(BATCH * HEADS, SEQ / 128), ablk(128);
  attn_fwd_wmma<<<agrid, ablk, 0, stream>>>(Qh, Kh, Vth, AO);

  // 4) output projection (f32 result)
  gemm_bf16_wmma<<<ggrid, gblk, 0, stream>>>(AO, Wot, bo, out, MROWS, EMBED, EMBED, 3, 1.0f);
}
